// AdaptiveProteinBlock_18270790877388
// MI455X (gfx1250) — matrix operand order repaired and checked by reading, not compiled, not run on hardware
//
#include <hip/hip_runtime.h>
#include <hip/hip_bf16.h>
#include <math.h>

// ---------------------------------------------------------------------------
// CDNA5 / gfx1250 implementation. wave32, WMMA bf16 16x16x32 with f32 accum.
// ---------------------------------------------------------------------------

typedef __bf16 bf16_t;
typedef __attribute__((ext_vector_type(16))) __bf16 v16bf;
typedef __attribute__((ext_vector_type(8)))  float  v8f;

#define NROW 4096
#define DIN  512
#define DATT 64
#define TOPP 0.9f

__device__ __forceinline__ v8f wmma_bf16(v16bf a, v16bf b, v8f c) {
  // v_wmma_f32_16x16x32_bf16 : D = A(16x32) * B(32x16) + C(16x16 f32)
  return __builtin_amdgcn_wmma_f32_16x16x32_bf16(
      /*neg_a=*/false, a, /*neg_b=*/false, b,
      /*c_mod=*/(short)0, c, /*reuse_a=*/false, /*reuse_b=*/false);
}

// A fragment: 16x32 bf16 tile from row-major [.,ld] storage.
// Lane l: row m = l&15; half = l>>4. VGPR v holds K pair per ISA 7.12.2 table.
__device__ __forceinline__ v16bf load_a_frag(const bf16_t* base, int ld,
                                             int row0, int k0, int lane) {
  int m = lane & 15, half = lane >> 4;
  const bf16_t* r = base + (size_t)(row0 + m) * ld + k0;
  v16bf a;
#pragma unroll
  for (int v = 0; v < 8; ++v) {
    int kk = (v < 4) ? (2 * v + 8 * half) : (16 + 2 * (v - 4) + 8 * half);
    a[2 * v]     = r[kk];
    a[2 * v + 1] = r[kk + 1];
  }
  return a;
}

// A fragment with clamped row shift (for the Rf/Rb chain priors).
__device__ __forceinline__ v16bf load_a_frag_shift(const bf16_t* base, int ld,
                                                   int row0, int shift,
                                                   int k0, int lane) {
  int m = lane & 15, half = lane >> 4;
  int row = row0 + m + shift;
  row = row < 0 ? 0 : (row > NROW - 1 ? NROW - 1 : row);
  const bf16_t* r = base + (size_t)row * ld + k0;
  v16bf a;
#pragma unroll
  for (int v = 0; v < 8; ++v) {
    int kk = (v < 4) ? (2 * v + 8 * half) : (16 + 2 * (v - 4) + 8 * half);
    a[2 * v]     = r[kk];
    a[2 * v + 1] = r[kk + 1];
  }
  return a;
}

// B fragment (32x16 = KxN) where memory holds Bt row-major [N][K] (weight
// style W[n][k], so B = W^T). Lane l: col n = l&15; half = l>>4.
// VGPR v holds K pair per ISA 7.12.2 table (same as A). Two 8-element runs ->
// compiler merges into global_load_b128.
__device__ __forceinline__ v16bf load_b_from_nk(const bf16_t* W, int ldk,
                                                int n0, int k0, int lane) {
  int n = lane & 15, half = lane >> 4;
  const bf16_t* r = W + (size_t)(n0 + n) * ldk + k0 + 8 * half;
  v16bf b;
#pragma unroll
  for (int v = 0; v < 8; ++v) {
    int kk = (v < 4) ? (2 * v) : (16 + 2 * (v - 4));
    b[2 * v]     = r[kk];
    b[2 * v + 1] = r[kk + 1];
  }
  return b;
}

// C/D 16x16 f32: lane n = l&15; VGPR v holds row m = v + 8*(l>>4).
__device__ __forceinline__ void store_c_f32(float* D, int ld, int row0,
                                            int col0, int lane, v8f c) {
  int n = lane & 15, half = lane >> 4;
#pragma unroll
  for (int v = 0; v < 8; ++v)
    D[(size_t)(row0 + v + 8 * half) * ld + col0 + n] = c[v];
}

__device__ __forceinline__ void store_c_bf16(bf16_t* D, int ld, int row0,
                                             int col0, int lane, v8f c) {
  int n = lane & 15, half = lane >> 4;
#pragma unroll
  for (int v = 0; v < 8; ++v)
    D[(size_t)(row0 + v + 8 * half) * ld + col0 + n] = (bf16_t)c[v];
}

__device__ __forceinline__ void addto_c_f32(float* D, int ld, int row0,
                                            int col0, int lane, v8f c) {
  int n = lane & 15, half = lane >> 4;
#pragma unroll
  for (int v = 0; v < 8; ++v) {
    size_t idx = (size_t)(row0 + v + 8 * half) * ld + col0 + n;
    D[idx] += c[v];
  }
}

// ---------------------------------------------------------------------------
// Elementwise f32 -> bf16 cast
// ---------------------------------------------------------------------------
__global__ void cast_kernel(const float* __restrict__ src,
                            bf16_t* __restrict__ dst, int n) {
  int i = blockIdx.x * 256 + threadIdx.x;
  if (i < n) dst[i] = (bf16_t)src[i];
}

// ---------------------------------------------------------------------------
// f32 [NROW x DIN] -> bf16 row-major copy AND bf16 transposed copy
// dstT is [DIN][NROW].  32x32 tiles staged in LDS for coalesced writes.
// ---------------------------------------------------------------------------
__global__ __launch_bounds__(256) void transpose_cast_kernel(
    const float* __restrict__ src, bf16_t* __restrict__ dstRM,
    bf16_t* __restrict__ dstT) {
  __shared__ bf16_t tile[32][33];
  int r0 = blockIdx.x * 32, c0 = blockIdx.y * 32;
  int tx = threadIdx.x & 31, ty = threadIdx.x >> 5;
#pragma unroll
  for (int j = 0; j < 4; ++j) {
    int r = ty + 8 * j;
    bf16_t bv = (bf16_t)src[(size_t)(r0 + r) * DIN + c0 + tx];
    dstRM[(size_t)(r0 + r) * DIN + c0 + tx] = bv;
    tile[r][tx] = bv;
  }
  __syncthreads();
#pragma unroll
  for (int j = 0; j < 4; ++j) {
    int cc = ty + 8 * j;
    dstT[(size_t)(c0 + cc) * NROW + r0 + tx] = tile[tx][cc];
  }
}

// ---------------------------------------------------------------------------
// q,k projections:  h = tanh(Xa W1^T); q = h W2^T; k = Xa W3^T.
// One block = 16 rows. 8 waves: waves 0-3 compute h (then q), waves 4-7 k.
// ---------------------------------------------------------------------------
__global__ __launch_bounds__(256) void qk_kernel(
    const bf16_t* __restrict__ Xa, const bf16_t* __restrict__ W1,
    const bf16_t* __restrict__ W2, const bf16_t* __restrict__ W3,
    bf16_t* __restrict__ qbf, bf16_t* __restrict__ kbf) {
  __shared__ bf16_t hTile[16 * DATT];
  int wave = threadIdx.x >> 5, lane = threadIdx.x & 31;
  int row0 = blockIdx.x * 16;
  int ncol = (wave & 3) * 16;
  bool isK = wave >= 4;
  const bf16_t* W = isK ? W3 : W1;

  v8f c = (v8f)(0.0f);
#pragma unroll 4
  for (int k0 = 0; k0 < DIN; k0 += 32) {
    v16bf a = load_a_frag(Xa, DIN, row0, k0, lane);
    v16bf b = load_b_from_nk(W, DIN, ncol, k0, lane);
    c = wmma_bf16(a, b, c);
  }
  if (isK) {
    store_c_bf16(kbf, DATT, row0, ncol, lane, c);
  } else {
    int n = lane & 15, half = lane >> 4;
#pragma unroll
    for (int v = 0; v < 8; ++v)
      hTile[(v + 8 * half) * DATT + ncol + n] = (bf16_t)tanhf(c[v]);
  }
  __syncthreads();
  if (!isK) {
    v8f cq = (v8f)(0.0f);
#pragma unroll
    for (int k0 = 0; k0 < DATT; k0 += 32) {
      v16bf a = load_a_frag(hTile, DATT, 0, k0, lane);
      v16bf b = load_b_from_nk(W2, DATT, ncol, k0, lane);
      cq = wmma_bf16(a, b, cq);
    }
    store_c_bf16(qbf, DATT, row0, ncol, lane, cq);
  }
}

// ---------------------------------------------------------------------------
// scores = q k^T  (TAU == 1).  Block = 16 rows x 512 cols; each wave owns
// four 16x16 tiles and reuses the q A-fragments across them. grid (256, 8).
// ---------------------------------------------------------------------------
__global__ __launch_bounds__(256) void scores_kernel(
    const bf16_t* __restrict__ q, const bf16_t* __restrict__ k,
    float* __restrict__ S) {
  int wave = threadIdx.x >> 5, lane = threadIdx.x & 31;
  int row0 = blockIdx.x * 16;
  int colBase = blockIdx.y * 512 + wave * 64;
  v16bf a0 = load_a_frag(q, DATT, row0, 0, lane);
  v16bf a1 = load_a_frag(q, DATT, row0, 32, lane);
#pragma unroll
  for (int t = 0; t < 4; ++t) {
    int col0 = colBase + t * 16;
    v8f c = (v8f)(0.0f);
    c = wmma_bf16(a0, load_b_from_nk(k, DATT, col0, 0, lane), c);
    c = wmma_bf16(a1, load_b_from_nk(k, DATT, col0, 32, lane), c);
    store_c_f32(S, NROW, row0, col0, lane, c);
  }
}

// ---------------------------------------------------------------------------
// Per-row softmax + nucleus(top-p) via threshold bisection + renorm -> bf16 A.
// keep_j  <=>  sum_{p > p_j} p < TOPP   (threshold set; no sort needed).
// ---------------------------------------------------------------------------
__device__ __forceinline__ float block_sum(float v, float* red) {
  int tid = threadIdx.x;
  red[tid] = v;
  __syncthreads();
  for (int s = 128; s > 0; s >>= 1) {
    if (tid < s) red[tid] += red[tid + s];
    __syncthreads();
  }
  float r = red[0];
  __syncthreads();
  return r;
}

__global__ __launch_bounds__(256) void nucleus_kernel(
    const float* __restrict__ S, bf16_t* __restrict__ Abf) {
  __shared__ float p[NROW];
  __shared__ float red[256];
  int row = blockIdx.x, tid = threadIdx.x;
  const float* srow = S + (size_t)row * NROW;

  float mx = -3.0e38f;
  for (int j = tid; j < NROW; j += 256) {
    float v = srow[j];
    p[j] = v;
    mx = fmaxf(mx, v);
  }
  red[tid] = mx;
  __syncthreads();
  for (int s = 128; s > 0; s >>= 1) {
    if (tid < s) red[tid] = fmaxf(red[tid], red[tid + s]);
    __syncthreads();
  }
  mx = red[0];
  __syncthreads();

  float lsum = 0.0f;
  for (int j = tid; j < NROW; j += 256) {
    float e = __expf(p[j] - mx);
    p[j] = e;
    lsum += e;
  }
  float total = block_sum(lsum, red);
  float inv = 1.0f / total;
  for (int j = tid; j < NROW; j += 256) p[j] *= inv;
  __syncthreads();

  // bisection for nucleus threshold: G(t) = sum of probs strictly > t
  float lo = 0.0f, hi = inv;  // max prob == exp(0)/total == inv
  for (int it = 0; it < 32; ++it) {
    float mid = 0.5f * (lo + hi);
    float g = 0.0f;
    for (int j = tid; j < NROW; j += 256) {
      float v = p[j];
      if (v > mid) g += v;
    }
    g = block_sum(g, red);
    if (g >= TOPP) lo = mid; else hi = mid;
  }

  float ms = 0.0f;
  for (int j = tid; j < NROW; j += 256) {
    float v = p[j];
    if (v > lo) ms += v;
  }
  ms = block_sum(ms, red);
  float invM = 1.0f / ms;
  bf16_t* arow = Abf + (size_t)row * NROW;
  for (int j = tid; j < NROW; j += 256) {
    float v = p[j];
    arow[j] = (bf16_t)((v > lo) ? v * invM : 0.0f);
  }
}

// ---------------------------------------------------------------------------
// XaNew = A(4096x4096) * Xa(4096x512).  Block = 16 rows x 512 cols; each
// wave owns 4 C tiles sharing one A fragment per K-step. B from XaT (bf16,
// [DIN][NROW]) -> contiguous b128 loads.  grid 256 blocks.
// ---------------------------------------------------------------------------
__global__ __launch_bounds__(256) void attnv_kernel(
    const bf16_t* __restrict__ A, const bf16_t* __restrict__ XaT,
    float* __restrict__ Out) {
  int wave = threadIdx.x >> 5, lane = threadIdx.x & 31;
  int row0 = blockIdx.x * 16;
  int colBase = wave * 64;
  v8f acc[4];
#pragma unroll
  for (int t = 0; t < 4; ++t) acc[t] = (v8f)(0.0f);

  const bf16_t* pfA = A + (size_t)(row0 + (lane & 15)) * NROW;
  for (int k0 = 0; k0 < NROW; k0 += 32) {
    if (k0 + 1024 < NROW)
      __builtin_prefetch((const void*)(pfA + k0 + 1024), 0, 0);
    v16bf a = load_a_frag(A, NROW, row0, k0, lane);
#pragma unroll
    for (int t = 0; t < 4; ++t) {
      v16bf b = load_b_from_nk(XaT, NROW, colBase + t * 16, k0, lane);
      acc[t] = wmma_bf16(a, b, acc[t]);
    }
  }
#pragma unroll
  for (int t = 0; t < 4; ++t)
    store_c_f32(Out, DIN, row0, colBase + t * 16, lane, acc[t]);
}

// ---------------------------------------------------------------------------
// Z += Xf U1^T + Xb U2^T + Xa U3^T  (Xf/Xb are clamped shifts of original X)
// Block = 16 rows x 512 cols; each wave 4 C tiles sharing 3 A fragments.
// ---------------------------------------------------------------------------
__global__ __launch_bounds__(256) void pf_kernel(
    const bf16_t* __restrict__ Xbf, const bf16_t* __restrict__ XaBf,
    const bf16_t* __restrict__ U1, const bf16_t* __restrict__ U2,
    const bf16_t* __restrict__ U3, float* __restrict__ Z, int shift) {
  int wave = threadIdx.x >> 5, lane = threadIdx.x & 31;
  int row0 = blockIdx.x * 16;
  int colBase = wave * 64;
  v8f acc[4];
#pragma unroll
  for (int t = 0; t < 4; ++t) acc[t] = (v8f)(0.0f);

  for (int k0 = 0; k0 < DIN; k0 += 32) {
    v16bf af = load_a_frag_shift(Xbf, DIN, row0, -shift, k0, lane);
    v16bf ab = load_a_frag_shift(Xbf, DIN, row0, +shift, k0, lane);
    v16bf aa = load_a_frag(XaBf, DIN, row0, k0, lane);
#pragma unroll
    for (int t = 0; t < 4; ++t) {
      int col0 = colBase + t * 16;
      acc[t] = wmma_bf16(af, load_b_from_nk(U1, DIN, col0, k0, lane), acc[t]);
      acc[t] = wmma_bf16(ab, load_b_from_nk(U2, DIN, col0, k0, lane), acc[t]);
      acc[t] = wmma_bf16(aa, load_b_from_nk(U3, DIN, col0, k0, lane), acc[t]);
    }
  }
#pragma unroll
  for (int t = 0; t < 4; ++t)
    addto_c_f32(Z, DIN, row0, colBase + t * 16, lane, acc[t]);
}

// ---------------------------------------------------------------------------
// y = X + Z ; LayerNorm(y) * gamma + beta.  One block per row, 512 cols.
// ---------------------------------------------------------------------------
__global__ __launch_bounds__(256) void ln_kernel(
    const float* __restrict__ X, const float* __restrict__ Z,
    const float* __restrict__ gamma, const float* __restrict__ beta,
    float* __restrict__ out) {
  __shared__ float red[256];
  int row = blockIdx.x, tid = threadIdx.x;
  size_t base = (size_t)row * DIN;
  float y0 = X[base + tid] + Z[base + tid];
  float y1 = X[base + tid + 256] + Z[base + tid + 256];
  float mu = block_sum(y0 + y1, red) * (1.0f / DIN);
  float d0 = y0 - mu, d1 = y1 - mu;
  float var = block_sum(d0 * d0 + d1 * d1, red) * (1.0f / DIN);
  float r = rsqrtf(var + 1e-5f);
  out[base + tid]       = d0 * r * gamma[tid] + beta[tid];
  out[base + tid + 256] = d1 * r * gamma[tid + 256] + beta[tid + 256];
}

// ---------------------------------------------------------------------------
// Host-side orchestration
// ---------------------------------------------------------------------------
static inline size_t align256(size_t x) { return (x + 255) & ~(size_t)255; }

extern "C" void kernel_launch(void* const* d_in, const int* in_sizes, int n_in,
                              void* d_out, int out_size, void* d_ws,
                              size_t ws_size, hipStream_t stream) {
  (void)in_sizes; (void)n_in; (void)out_size; (void)ws_size;
  const float* X     = (const float*)d_in[0];
  const float* W1    = (const float*)d_in[1];
  const float* W2    = (const float*)d_in[2];
  const float* W3    = (const float*)d_in[3];
  const float* U1    = (const float*)d_in[4];
  const float* U2    = (const float*)d_in[5];
  const float* U3    = (const float*)d_in[6];
  const float* gamma = (const float*)d_in[7];
  const float* beta  = (const float*)d_in[8];
  float* out = (float*)d_out;

  char* ws = (char*)d_ws;
  size_t off = 0;
  auto take = [&](size_t bytes) { char* p = ws + off; off += align256(bytes); return p; };

  bf16_t* Xbf   = (bf16_t*)take((size_t)NROW * DIN * 2);
  bf16_t* XaBf  = (bf16_t*)take((size_t)NROW * DIN * 2);
  bf16_t* XaT   = (bf16_t*)take((size_t)DIN * NROW * 2);
  bf16_t* W1bf  = (bf16_t*)take((size_t)DATT * DIN * 2);
  bf16_t* W2bf  = (bf16_t*)take((size_t)DATT * DATT * 2);
  bf16_t* W3bf  = (bf16_t*)take((size_t)DATT * DIN * 2);
  bf16_t* U1bf  = (bf16_t*)take((size_t)2 * DIN * DIN * 2);
  bf16_t* U2bf  = (bf16_t*)take((size_t)2 * DIN * DIN * 2);
  bf16_t* U3bf  = (bf16_t*)take((size_t)2 * DIN * DIN * 2);
  bf16_t* qbf   = (bf16_t*)take((size_t)NROW * DATT * 2);
  bf16_t* kbf   = (bf16_t*)take((size_t)NROW * DATT * 2);
  float*  S     = (float*) take((size_t)NROW * NROW * 4);
  bf16_t* Abf   = (bf16_t*)take((size_t)NROW * NROW * 2);
  float*  XaNew = (float*) take((size_t)NROW * DIN * 4);
  float*  Z     = (float*) take((size_t)NROW * DIN * 4);

  auto cast = [&](const float* s, bf16_t* d, int n) {
    cast_kernel<<<(n + 255) / 256, 256, 0, stream>>>(s, d, n);
  };

  // One-time casts (re-done every call for determinism)
  cast(X, Xbf, NROW * DIN);
  transpose_cast_kernel<<<dim3(NROW / 32, DIN / 32), 256, 0, stream>>>(
      X, XaBf, XaT);
  cast(W1, W1bf, DATT * DIN);
  cast(W2, W2bf, DATT * DATT);
  cast(W3, W3bf, DATT * DIN);
  cast(U1, U1bf, 2 * DIN * DIN);
  cast(U2, U2bf, 2 * DIN * DIN);
  cast(U3, U3bf, 2 * DIN * DIN);
  hipMemsetAsync(Z, 0, (size_t)NROW * DIN * 4, stream);

  for (int n = 0; n < 2; ++n) {
    qk_kernel<<<NROW / 16, 256, 0, stream>>>(XaBf, W1bf, W2bf, W3bf, qbf, kbf);
    scores_kernel<<<dim3(NROW / 16, NROW / 512), 256, 0, stream>>>(qbf, kbf, S);
    nucleus_kernel<<<NROW, 256, 0, stream>>>(S, Abf);
    attnv_kernel<<<NROW / 16, 256, 0, stream>>>(Abf, XaT, XaNew);
    transpose_cast_kernel<<<dim3(NROW / 32, DIN / 32), 256, 0, stream>>>(
        XaNew, XaBf, XaT);
    pf_kernel<<<NROW / 16, 256, 0, stream>>>(
        Xbf, XaBf, U1bf + (size_t)n * DIN * DIN, U2bf + (size_t)n * DIN * DIN,
        U3bf + (size_t)n * DIN * DIN, Z, n + 1);
  }
  ln_kernel<<<NROW, 256, 0, stream>>>(X, Z, gamma, beta, out);
}
